// CausalSelfAttention_51410758533254
// MI455X (gfx1250) — matrix operand order repaired and checked by reading, not compiled, not run on hardware
//
#include <hip/hip_runtime.h>
#include <hip/hip_bf16.h>

// Problem constants (match reference)
static constexpr int BATCH = 4;
static constexpr int TSEQ  = 2048;
static constexpr int CDIM  = 1024;
static constexpr int NH    = 16;
static constexpr int DH    = 64;

typedef __attribute__((ext_vector_type(16))) __bf16 bf16x16;
typedef __attribute__((ext_vector_type(8)))  __bf16 bf16x8;
typedef __attribute__((ext_vector_type(4)))  __bf16 bf16x4;
typedef __attribute__((ext_vector_type(8)))  float  f32x8;

union Frag {
  bf16x16 v;
  bf16x8  h[2];
};

static __device__ __forceinline__ f32x8 wmma_bf16(const Frag& a, const Frag& b, f32x8 c) {
  // D = A(16x32 bf16) * B(32x16 bf16) + C(16x16 f32)
  return __builtin_amdgcn_wmma_f32_16x16x32_bf16(false, a.v, false, b.v, (short)0, c, false, false);
}

// Async global -> LDS copy of 16 bytes per lane (ASYNCcnt-tracked, no VGPR data path)
static __device__ __forceinline__ void async_copy_b128(uint32_t lds_off, int voff,
                                                       unsigned long long sbase) {
  asm volatile("global_load_async_to_lds_b128 %0, %1, %2"
               :: "v"(lds_off), "v"(voff), "s"(sbase) : "memory");
}

// ---------------------------------------------------------------------------
// Pre-pass: f32 -> bf16 elementwise
// ---------------------------------------------------------------------------
__global__ __launch_bounds__(256)
void convert_f32_bf16_kernel(const float* __restrict__ in, __bf16* __restrict__ out,
                             int n4 /* count of float4 */)
{
  int i = blockIdx.x * blockDim.x + threadIdx.x;
  if (i < n4) {
    const float4 f = *(const float4*)(in + (size_t)i * 4);
    bf16x4 h = { (__bf16)f.x, (__bf16)f.y, (__bf16)f.z, (__bf16)f.w };
    *(bf16x4*)(out + (size_t)i * 4) = h;
  }
}

// ---------------------------------------------------------------------------
// Pre-pass: W[K,N] f32 -> Wt[N,K] bf16 (transpose through LDS, 64x64 tiles)
// ---------------------------------------------------------------------------
__global__ __launch_bounds__(256)
void transpose_w_kernel(const float* __restrict__ W, __bf16* __restrict__ Wt,
                        int K, int N)
{
  __shared__ float Ts[64][65];
  const int tid = threadIdx.x;
  const int n0 = blockIdx.x * 64;
  const int k0 = blockIdx.y * 64;

  #pragma unroll
  for (int i = 0; i < 4; i++) {
    int idx = i * 256 + tid;       // 0..1023 float4s of a 64x64 tile
    int k  = idx >> 4;
    int c4 = (idx & 15) << 2;
    const float4 g = *(const float4*)(W + (size_t)(k0 + k) * N + n0 + c4);
    Ts[c4 + 0][k] = g.x;
    Ts[c4 + 1][k] = g.y;
    Ts[c4 + 2][k] = g.z;
    Ts[c4 + 3][k] = g.w;
  }
  __syncthreads();
  #pragma unroll
  for (int i = 0; i < 4; i++) {
    int idx = i * 256 + tid;
    int n  = idx >> 4;
    int c4 = (idx & 15) << 2;
    bf16x4 h = { (__bf16)Ts[n][c4 + 0], (__bf16)Ts[n][c4 + 1],
                 (__bf16)Ts[n][c4 + 2], (__bf16)Ts[n][c4 + 3] };
    *(bf16x4*)(Wt + (size_t)(n0 + n) * K + k0 + c4) = h;
  }
}

// ---------------------------------------------------------------------------
// GEMM: C[M,N] = A[M,K](bf16,row-major) @ Bt[N,K](bf16,row-major) + bias[N]
// Async-to-LDS double-buffered pipeline, block tile 256x128, BK=32.
// 8 waves in 4(M) x 2(N); each wave owns a 64x64 tile: 16 WMMAs / stage.
// mode 0: scatter into q / k / vT bf16 buffers.  mode 1: f32 out + bias.
// ---------------------------------------------------------------------------
__global__ __launch_bounds__(256)
void gemm_async_kernel(const __bf16* __restrict__ A, const __bf16* __restrict__ Bt,
                       const float* __restrict__ bias, int M, int N, int K,
                       int mode, __bf16* __restrict__ qb, __bf16* __restrict__ kb,
                       __bf16* __restrict__ vtb, float* __restrict__ outp)
{
  // 40 bf16 per row (32 + 8 pad) -> 80B row stride: every 16B chunk stays aligned
  __shared__ alignas(16) __bf16 As[2][256][40];   // 40 KB
  __shared__ alignas(16) __bf16 Bs[2][128][40];   // 20 KB

  const int tid  = threadIdx.x;
  const int wv   = tid >> 5;
  const int lane = tid & 31;
  const int l16  = lane & 15;
  const int hi   = lane >> 4;
  const int wm   = wv >> 1;   // 0..3 : 64-row slab
  const int wn   = wv & 1;    // 0..1 : 64-col slab
  const int m0   = blockIdx.y * 256;
  const int n0   = blockIdx.x * 128;

  const uint32_t ldsA = (uint32_t)(uintptr_t)&As[0][0][0];
  const uint32_t ldsB = (uint32_t)(uintptr_t)&Bs[0][0][0];
  const unsigned long long aBase = (unsigned long long)(uintptr_t)A;
  const unsigned long long bBase = (unsigned long long)(uintptr_t)Bt;
  constexpr int STAGE_A = 256 * 80;   // bytes per A stage
  constexpr int STAGE_B = 128 * 80;   // bytes per B stage

  // A tile: 256x32 bf16 = 1024 16B chunks (4/thread); B tile: 128x32 = 512 (2/thread)
  auto issue_stage = [&](int k0, int buf) {
    #pragma unroll
    for (int j = 0; j < 4; j++) {
      const int idx = j * 256 + tid;
      const int row = idx >> 2, ch = idx & 3;
      async_copy_b128(ldsA + buf * STAGE_A + row * 80 + ch * 16,
                      (int)((((size_t)(m0 + row)) * K + k0 + ch * 8) * 2), aBase);
    }
    #pragma unroll
    for (int j = 0; j < 2; j++) {
      const int idx = j * 256 + tid;
      const int row = idx >> 2, ch = idx & 3;
      async_copy_b128(ldsB + buf * STAGE_B + row * 80 + ch * 16,
                      (int)((((size_t)(n0 + row)) * K + k0 + ch * 8) * 2), bBase);
    }
  };

  f32x8 acc[4][4];
  #pragma unroll
  for (int mi = 0; mi < 4; mi++)
    #pragma unroll
    for (int ni = 0; ni < 4; ni++)
      acc[mi][ni] = (f32x8){0,0,0,0,0,0,0,0};

  auto compute_stage = [&](int buf) {
    Frag a[4], b[4];
    #pragma unroll
    for (int mi = 0; mi < 4; mi++) {
      const __bf16* rp = &As[buf][wm * 64 + mi * 16 + l16][0];
      a[mi].h[0] = *(const bf16x8*)(rp + hi * 16);
      a[mi].h[1] = *(const bf16x8*)(rp + hi * 16 + 8);
    }
    #pragma unroll
    for (int ni = 0; ni < 4; ni++) {
      const __bf16* cp = &Bs[buf][wn * 64 + ni * 16 + l16][0];
      b[ni].h[0] = *(const bf16x8*)(cp + hi * 16);
      b[ni].h[1] = *(const bf16x8*)(cp + hi * 16 + 8);
    }
    #pragma unroll
    for (int mi = 0; mi < 4; mi++)
      #pragma unroll
      for (int ni = 0; ni < 4; ni++)
        acc[mi][ni] = wmma_bf16(a[mi], b[ni], acc[mi][ni]);
  };

  issue_stage(0, 0);

  int k0 = 0;
  for (; k0 < K - 32; k0 += 32) {           // steady state: branch-free body
    const int buf = (k0 >> 5) & 1;
    issue_stage(k0 + 32, buf ^ 1);
    // 6 newer copies may remain in flight; the 6 for `buf` must be done
    asm volatile("s_wait_asynccnt 0x6" ::: "memory");
    __syncthreads();
    compute_stage(buf);
    __syncthreads();                        // done reading `buf` before refill
  }
  { // final (peeled) stage
    const int buf = (k0 >> 5) & 1;
    asm volatile("s_wait_asynccnt 0x0" ::: "memory");
    __syncthreads();
    compute_stage(buf);
  }

  // Epilogue. C/D layout: lane holds (row = r + 8*hi, col = l16) in VGPR r.
  #pragma unroll
  for (int mi = 0; mi < 4; mi++) {
    #pragma unroll
    for (int ni = 0; ni < 4; ni++) {
      const int gcol = n0 + wn * 64 + ni * 16 + l16;
      const float bv = bias[gcol];
      #pragma unroll
      for (int r = 0; r < 8; r++) {
        const int grow = m0 + wm * 64 + mi * 16 + r + hi * 8;
        const float val = acc[mi][ni][r] + bv;
        if (mode == 0) {
          const int bb = grow >> 11;          // / TSEQ
          const int t  = grow & (TSEQ - 1);
          if (gcol < CDIM) {                  // Q -> [B,H,T,DH]
            const int h = gcol >> 6, d = gcol & 63;
            qb[((size_t)(bb * NH + h) * TSEQ + t) * DH + d] = (__bf16)val;
          } else if (gcol < 2 * CDIM) {       // K -> [B,H,T,DH]
            const int c = gcol - CDIM;
            const int h = c >> 6, d = c & 63;
            kb[((size_t)(bb * NH + h) * TSEQ + t) * DH + d] = (__bf16)val;
          } else {                            // V -> transposed [B,H,DH,T]
            const int c = gcol - 2 * CDIM;
            const int h = c >> 6, d = c & 63;
            vtb[((size_t)(bb * NH + h) * DH + d) * TSEQ + t] = (__bf16)val;
          }
        } else {
          outp[(size_t)grow * N + gcol] = val;
        }
      }
    }
  }
}

// ---------------------------------------------------------------------------
// Flash attention: grid (B*H, T/64), block 128 (4 waves).
// Each wave owns 16 query rows; loops key tiles of 32 with online softmax.
// q,k: bf16 [B,H,T,64]; vt: bf16 [B,H,64,T]; y: bf16 [B,T,C].
// ---------------------------------------------------------------------------
__global__ __launch_bounds__(128)
void flash_attn_kernel(const __bf16* __restrict__ q, const __bf16* __restrict__ kmat,
                       const __bf16* __restrict__ vt, __bf16* __restrict__ y)
{
  __shared__ alignas(16) __bf16 Pt[4][16][40];   // per-wave P staging tile

  const int tid  = threadIdx.x;
  const int wv   = tid >> 5;
  const int lane = tid & 31;
  const int l16  = lane & 15;
  const int hi   = lane >> 4;

  const int bh = blockIdx.x;                 // 0..63
  const int q0 = blockIdx.y * 64 + wv * 16;  // first query row of this wave

  const __bf16* qp = q    + (size_t)bh * TSEQ * DH;
  const __bf16* kp = kmat + (size_t)bh * TSEQ * DH;
  const __bf16* vp = vt   + (size_t)bh * DH * TSEQ;

  // Q fragments: two K-chunks of 32 over DH=64, loaded straight from HBM
  Frag aq[2];
  {
    const __bf16* rp = qp + (size_t)(q0 + l16) * DH;
    #pragma unroll
    for (int c = 0; c < 2; c++) {
      const int kk = c * 32;
      aq[c].h[0] = *(const bf16x8*)(rp + kk + hi * 16);
      aq[c].h[1] = *(const bf16x8*)(rp + kk + hi * 16 + 8);
    }
  }

  f32x8 accy[4];
  #pragma unroll
  for (int t = 0; t < 4; t++) accy[t] = (f32x8){0,0,0,0,0,0,0,0};
  float mrow[8], lrow[8];
  #pragma unroll
  for (int r = 0; r < 8; r++) { mrow[r] = -1e30f; lrow[r] = 0.0f; }

  const int kend = q0 + 16;                  // causal: keys [0, q0+16)
  for (int k0 = 0; k0 < kend; k0 += 32) {
    // S = Q @ K^T for keys [k0, k0+32): two 16x16 frags
    f32x8 s0 = (f32x8){0,0,0,0,0,0,0,0};
    f32x8 s1 = (f32x8){0,0,0,0,0,0,0,0};
    #pragma unroll
    for (int c = 0; c < 2; c++) {
      const int kk = c * 32;
      Frag b0, b1;
      const __bf16* kr0 = kp + (size_t)(k0 + l16) * DH + kk + hi * 16;
      b0.h[0] = *(const bf16x8*)(kr0);
      b0.h[1] = *(const bf16x8*)(kr0 + 8);
      const __bf16* kr1 = kp + (size_t)(k0 + 16 + l16) * DH + kk + hi * 16;
      b1.h[0] = *(const bf16x8*)(kr1);
      b1.h[1] = *(const bf16x8*)(kr1 + 8);
      s0 = wmma_bf16(aq[c], b0, s0);
      s1 = wmma_bf16(aq[c], b1, s1);
    }

    // Scale 1/sqrt(64) and causal mask (diagonal-overlapping tiles only)
    const float scale = 0.125f;
    const bool needMask = (k0 + 31 > q0);
    #pragma unroll
    for (int r = 0; r < 8; r++) {
      const int qrow = q0 + r + hi * 8;
      float v0 = s0[r] * scale;
      float v1 = s1[r] * scale;
      if (needMask) {
        if (k0 + l16 > qrow)      v0 = -1e30f;
        if (k0 + 16 + l16 > qrow) v1 = -1e30f;
      }
      s0[r] = v0; s1[r] = v1;
    }

    // Online softmax: row max/sum across 16-lane halves (rows r / r+8)
    float alpha[8];
    #pragma unroll
    for (int r = 0; r < 8; r++) {
      float tmax = fmaxf(s0[r], s1[r]);
      #pragma unroll
      for (int msk = 1; msk < 16; msk <<= 1) tmax = fmaxf(tmax, __shfl_xor(tmax, msk, 32));
      const float mn = fmaxf(mrow[r], tmax);
      alpha[r] = __expf(mrow[r] - mn);
      mrow[r] = mn;
      const float p0 = __expf(s0[r] - mn);
      const float p1 = __expf(s1[r] - mn);
      s0[r] = p0; s1[r] = p1;
      float rs = p0 + p1;
      #pragma unroll
      for (int msk = 1; msk < 16; msk <<= 1) rs += __shfl_xor(rs, msk, 32);
      lrow[r] = lrow[r] * alpha[r] + rs;
    }
    #pragma unroll
    for (int t = 0; t < 4; t++)
      #pragma unroll
      for (int r = 0; r < 8; r++) accy[t][r] *= alpha[r];

    // D-layout -> A-layout relayout of P through per-wave LDS tile
    #pragma unroll
    for (int r = 0; r < 8; r++) {
      const int row = r + hi * 8;
      Pt[wv][row][l16]      = (__bf16)s0[r];
      Pt[wv][row][l16 + 16] = (__bf16)s1[r];
    }
    asm volatile("s_wait_dscnt 0" ::: "memory");  // DS in-order per wave
    Frag pf;
    {
      const __bf16* rp = &Pt[wv][l16][0];
      pf.h[0] = *(const bf16x8*)(rp + hi * 16);
      pf.h[1] = *(const bf16x8*)(rp + hi * 16 + 8);
    }

    // Y += P @ V  (V^T rows are contiguous columns of V)
    #pragma unroll
    for (int t = 0; t < 4; t++) {
      Frag bv;
      const __bf16* vr = vp + (size_t)(t * 16 + l16) * TSEQ + k0 + hi * 16;
      bv.h[0] = *(const bf16x8*)(vr);
      bv.h[1] = *(const bf16x8*)(vr + 8);
      accy[t] = wmma_bf16(pf, bv, accy[t]);
    }
    asm volatile("" ::: "memory");
  }

  // Normalize and store y[b, t, h*64+d] as bf16 (feeds async-GEMM)
  const int bb = bh >> 4;
  const int h  = bh & 15;
  #pragma unroll
  for (int t = 0; t < 4; t++) {
    #pragma unroll
    for (int r = 0; r < 8; r++) {
      const int trow = q0 + r + hi * 8;
      const int d    = t * 16 + l16;
      y[((size_t)(bb * TSEQ + trow)) * CDIM + h * DH + d] =
          (__bf16)(accy[t][r] / lrow[r]);
    }
  }
}

// ---------------------------------------------------------------------------
extern "C" void kernel_launch(void* const* d_in, const int* in_sizes, int n_in,
                              void* d_out, int out_size, void* d_ws, size_t ws_size,
                              hipStream_t stream)
{
  (void)in_sizes; (void)n_in; (void)out_size; (void)ws_size;
  const float* x      = (const float*)d_in[0];
  const float* w_qkv  = (const float*)d_in[1];
  const float* b_qkv  = (const float*)d_in[2];
  const float* w_proj = (const float*)d_in[3];
  const float* b_proj = (const float*)d_in[4];
  float* out = (float*)d_out;

  char* ws = (char*)d_ws;
  const size_t MK   = (size_t)BATCH * TSEQ * CDIM;          // 8M elements
  const size_t qkvE = (size_t)BATCH * NH * TSEQ * DH;       // 8M elements
  __bf16* xb   = (__bf16*)(ws);                             // [M,K]        16MB
  __bf16* wqT  = (__bf16*)(ws + MK * 2);                    // [3C,K]        6MB
  __bf16* wpT  = (__bf16*)(ws + MK * 2 + (size_t)3 * CDIM * CDIM * 2);  // [C,K] 2MB
  char*   p    = ws + MK * 2 + (size_t)4 * CDIM * CDIM * 2;
  __bf16* qb   = (__bf16*)(p);
  __bf16* kb   = (__bf16*)(p + qkvE * 2);
  __bf16* vtb  = (__bf16*)(p + qkvE * 4);
  __bf16* yb   = (__bf16*)(p + qkvE * 6);                   // [M,C] bf16   16MB

  const int M = BATCH * TSEQ;   // 8192

  { // x -> bf16
    int n4 = (int)(MK / 4);
    convert_f32_bf16_kernel<<<(n4 + 255) / 256, 256, 0, stream>>>(x, xb, n4);
  }
  { // w_qkv [K,3C] -> [3C,K] bf16 ; w_proj [K,C] -> [C,K] bf16
    transpose_w_kernel<<<dim3(3 * CDIM / 64, CDIM / 64), 256, 0, stream>>>(w_qkv, wqT, CDIM, 3 * CDIM);
    transpose_w_kernel<<<dim3(CDIM / 64, CDIM / 64), 256, 0, stream>>>(w_proj, wpT, CDIM, CDIM);
  }
  { // QKV projection (async double-buffered GEMM, 256x128 tiles)
    dim3 grid((3 * CDIM) / 128, M / 256);
    gemm_async_kernel<<<grid, 256, 0, stream>>>(xb, wqT, b_qkv, M, 3 * CDIM, CDIM,
                                                0, qb, kb, vtb, nullptr);
  }
  { // Flash attention
    dim3 grid(BATCH * NH, TSEQ / 64);
    flash_attn_kernel<<<grid, 128, 0, stream>>>(qb, kb, vtb, yb);
  }
  { // Output projection
    dim3 grid(CDIM / 128, M / 256);
    gemm_async_kernel<<<grid, 256, 0, stream>>>(yb, wpT, b_proj, M, CDIM, CDIM,
                                                1, nullptr, nullptr, nullptr, out);
  }
}
